// TinyWaveRNN_25168508354652
// MI455X (gfx1250) — matrix operand order repaired and checked by reading, not compiled, not run on hardware
//
#include <hip/hip_runtime.h>

// ---------------------------------------------------------------------------
// TinyWaveRNN for MI455X (gfx1250, wave32):
//  - all matmuls on v_wmma_f32_16x16x32_f16 (f16 in / f32 accumulate)
//  - GEMM tiles staged with global_load_async_to_lds_b128 (ASYNCcnt),
//    double-buffered so DMA of step k+1 overlaps WMMA of step k
//  - inherently-sequential sparse GRU runs as 2 WGs x 16 waves with the
//    hidden state in LDS and 8 fully-unrolled WMMAs per 16x16 output tile
// ---------------------------------------------------------------------------

#define Bc   32
#define Tc   2048
#define BT   (Bc * Tc)          // 65536
#define Hc   256
#define G3   (3 * Hc)           // 768
#define GIN  97
#define GPAD 128                // gru_in padded K

typedef __attribute__((ext_vector_type(16))) _Float16 v16h;
typedef __attribute__((ext_vector_type(8)))  _Float16 v8h;
typedef __attribute__((ext_vector_type(8)))  float    v8f;

__device__ __forceinline__ float sigf(float x) { return 1.0f / (1.0f + __expf(-x)); }

// CDNA5 async global->LDS copy (ASYNCcnt-tracked); LDS offset = low 32 bits of
// the generic shared pointer (flat LDS aperture truncates to the DS offset).
__device__ __forceinline__ void async_ld_b128(const void* lds_ptr, const void* g) {
  unsigned loff = (unsigned)(uintptr_t)lds_ptr;
  asm volatile("global_load_async_to_lds_b128 %0, %1, off"
               :: "v"(loff), "v"((unsigned long long)(uintptr_t)g)
               : "memory");
}
__device__ __forceinline__ void wait_async0() {
  asm volatile("s_wait_asynccnt 0" ::: "memory");
}

// ---------------------------------------------------------------------------
// 1. feature embedding + concat
// ---------------------------------------------------------------------------
__global__ void embed_concat(const float* __restrict__ mel,
                             const float* __restrict__ f0,
                             const float* __restrict__ uv,
                             const float* __restrict__ f0w, const float* __restrict__ f0b,
                             const float* __restrict__ uvw, const float* __restrict__ uvb,
                             float* __restrict__ cin) {
  int bt = blockIdx.x * blockDim.x + threadIdx.x;
  if (bt >= BT) return;
  const float* m = mel + (size_t)bt * 80;
  float* o = cin + (size_t)bt * 112;
  #pragma unroll 8
  for (int i = 0; i < 80; ++i) o[i] = m[i];
  float fv = f0[bt], uvv = uv[bt];
  #pragma unroll
  for (int i = 0; i < 16; ++i) o[80 + i] = fv * f0w[i] + f0b[i];
  #pragma unroll
  for (int i = 0; i < 16; ++i) o[96 + i] = uvv * uvw[i] + uvb[i];
}

// ---------------------------------------------------------------------------
// 2. 3-tap conv (pad=1) + leaky_relu(0.1). in: [B,T,Cin], w: [Cout,Cin,3]
// ---------------------------------------------------------------------------
__global__ void conv1d_lrelu(const float* __restrict__ in,
                             const float* __restrict__ w,
                             const float* __restrict__ bias,
                             float* __restrict__ out, int Cin, int Cout) {
  int idx = blockIdx.x * blockDim.x + threadIdx.x;
  if (idx >= BT * Cout) return;
  int o  = idx % Cout;
  int bt = idx / Cout;
  int t  = bt % Tc;
  int b  = bt / Tc;
  float acc = bias[o];
  for (int k = 0; k < 3; ++k) {
    int tt = t + k - 1;
    if (tt < 0 || tt >= Tc) continue;
    const float* row = in + ((size_t)b * Tc + tt) * Cin;
    const float* wr  = w + (size_t)o * Cin * 3 + k;
    for (int i = 0; i < Cin; ++i) acc += row[i] * wr[(size_t)i * 3];
  }
  out[(size_t)bt * Cout + o] = acc > 0.0f ? acc : 0.1f * acc;
}

// ---------------------------------------------------------------------------
// 3. tiny conditioning GRU (hidden 32 per direction)
// ---------------------------------------------------------------------------
__global__ void cond_gru_scan(const float* __restrict__ x,   // [B,T,64]
                              const float* __restrict__ wih, // [96,64]
                              const float* __restrict__ whh, // [96,32]
                              const float* __restrict__ bih,
                              const float* __restrict__ bhh,
                              float* __restrict__ xcat,      // [B,T,64]
                              int rev, int coloff) {
  __shared__ float h[32];
  __shared__ float xt[64];
  int j = threadIdx.x;
  int b = blockIdx.x;
  h[j] = 0.0f;
  __syncthreads();
  for (int s = 0; s < Tc; ++s) {
    int t = rev ? (Tc - 1 - s) : s;
    const float* xr = x + ((size_t)b * Tc + t) * 64;
    xt[j] = xr[j]; xt[j + 32] = xr[j + 32];
    __syncthreads();
    float gi[3], gh[3];
    #pragma unroll
    for (int g = 0; g < 3; ++g) {
      int row = g * 32 + j;
      float a = bih[row];
      const float* wr = wih + (size_t)row * 64;
      for (int k = 0; k < 64; ++k) a += wr[k] * xt[k];
      gi[g] = a;
      float c = bhh[row];
      const float* hr = whh + (size_t)row * 32;
      for (int k = 0; k < 32; ++k) c += hr[k] * h[k];
      gh[g] = c;
    }
    float r = sigf(gi[0] + gh[0]);
    float z = sigf(gi[1] + gh[1]);
    float n = tanhf(gi[2] + r * gh[2]);
    float hn = (1.0f - z) * n + z * h[j];
    __syncthreads();
    h[j] = hn;
    xcat[((size_t)b * Tc + t) * 64 + coloff + j] = hn;
    __syncthreads();
  }
}

// ---------------------------------------------------------------------------
// 4. async-staged WMMA GEMM: out[M,N] = act([A1|A2] @ W^T + bias)
//    Requirements (guaranteed by all call sites): M%128==0, K%32==0,
//    N % (NSUB*16) == 0, split K1%32==0 (or K2==0).
//    Block = 256 thr (8 waves); macro-tile 128 x (NSUB*16); double-buffered
//    LDS tiles filled by global_load_async_to_lds_b128.
// ---------------------------------------------------------------------------
template <int NSUB>
__device__ __forceinline__ void stage_tiles(float* Asd, float* Wsd,
                                            const float* A1, int K1,
                                            const float* A2, int K2,
                                            const float* W, int K,
                                            int m_base, int n_base, int k0, int tid) {
  const float* Asrc = (k0 < K1) ? A1 : A2;           // scalar (wave-uniform) select
  const int    Kcur = (k0 < K1) ? K1 : K2;
  const int    krel = (k0 < K1) ? k0 : (k0 - K1);
  #pragma unroll
  for (int j = 0; j < 4; ++j) {                      // A: 128x32 f32 = 1024 b128
    int chunk = tid + j * 256;
    int row = chunk >> 3;
    int kq  = (chunk & 7) << 2;
    async_ld_b128(Asd + row * 32 + kq,
                  Asrc + (size_t)(m_base + row) * Kcur + krel + kq);
  }
  for (int chunk = tid; chunk < NSUB * 128; chunk += 256) {  // W: NSUB*16 x 32
    int row = chunk >> 3, kq = (chunk & 7) << 2;
    async_ld_b128(Wsd + row * 32 + kq,
                  W + (size_t)(n_base + row) * K + k0 + kq);
  }
}

template <int NSUB>
__global__ void wmma_gemm_fast(const float* __restrict__ A1, int K1,
                               const float* __restrict__ A2, int K2,
                               const float* __restrict__ W,
                               const float* __restrict__ bias,
                               float* __restrict__ out,
                               int M, int N, int act) {
  __shared__ __align__(16) float As[2][128 * 32];        // 2 x 16 KB
  __shared__ __align__(16) float Wsh[2][NSUB * 16 * 32]; // 2 x NSUB*2 KB
  const int K = K1 + K2;
  const int tid  = threadIdx.x;
  const int wave = tid >> 5, lane = tid & 31;
  const int l15  = lane & 15, hf = lane >> 4;
  const int m_base = blockIdx.y * 128;
  const int n_base = blockIdx.x * (NSUB * 16);

  v8f c[NSUB] = {};
  stage_tiles<NSUB>(As[0], Wsh[0], A1, K1, A2, K2, W, K, m_base, n_base, 0, tid);

  int buf = 0;
  for (int k0 = 0; k0 < K; k0 += 32, buf ^= 1) {
    wait_async0();                 // own async writes for buf are done
    __syncthreads();               // => everyone's are; prior reads of buf^1 done
    if (k0 + 32 < K)               // overlap: DMA next tile while computing
      stage_tiles<NSUB>(As[buf ^ 1], Wsh[buf ^ 1],
                        A1, K1, A2, K2, W, K, m_base, n_base, k0 + 32, tid);
    // pack fragments (contiguous LDS runs, branch-free)
    const float* ar = As[buf] + (wave * 16 + l15) * 32;
    v16h a;
    #pragma unroll
    for (int e = 0; e < 8; ++e) {
      a[e]     = (_Float16)ar[e + (hf << 4)];        // K 0..7  (+16*hf)
      a[e + 8] = (_Float16)ar[e + 8 + (hf << 4)];    // K 8..15 (+16*hf)
    }
    #pragma unroll
    for (int ns = 0; ns < NSUB; ++ns) {
      const float* br = Wsh[buf] + (ns * 16 + l15) * 32 + (hf << 4);
      v16h b;
      #pragma unroll
      for (int e = 0; e < 8; ++e) { b[e] = (_Float16)br[e]; b[e + 8] = (_Float16)br[e + 8]; }
      c[ns] = __builtin_amdgcn_wmma_f32_16x16x32_f16(false, a, false, b,
                                                     (short)0, c[ns], false, false);
    }
  }

  const int mrow_w = m_base + wave * 16;
  #pragma unroll
  for (int ns = 0; ns < NSUB; ++ns) {
    int ncol = n_base + ns * 16 + l15;
    float bb = bias[ncol];
    #pragma unroll
    for (int v = 0; v < 8; ++v) {                    // C/D: row = vgpr + 8*(lane>=16)
      float r = c[ns][v] + bb;
      if (act == 1) r = tanhf(r);
      out[(size_t)(mrow_w + v + (hf << 3)) * N + ncol] = r;
    }
  }
}

// ---------------------------------------------------------------------------
// 5. gru_in = [audio | cond | 0-pad] -> [BT, 128]
// ---------------------------------------------------------------------------
__global__ void build_gruin(const float* __restrict__ audio,
                            const float* __restrict__ cond,
                            float* __restrict__ gin) {
  int bt = blockIdx.x * blockDim.x + threadIdx.x;
  if (bt >= BT) return;
  float* o = gin + (size_t)bt * GPAD;
  o[0] = audio[bt];
  const float* cr = cond + (size_t)bt * 96;
  #pragma unroll 8
  for (int i = 0; i < 96; ++i) o[1 + i] = cr[i];
  #pragma unroll
  for (int i = GIN; i < GPAD; ++i) o[i] = 0.0f;
}

// ---------------------------------------------------------------------------
// 6. masked weights: wih*mask zero-padded to [768,128] f32; whh*mask -> f16
// ---------------------------------------------------------------------------
__global__ void mask_weights(const float* __restrict__ wih, const float* __restrict__ ihm,
                             const float* __restrict__ whh, const float* __restrict__ hhm,
                             float* __restrict__ wih_m, _Float16* __restrict__ whh_h) {
  int i = blockIdx.x * blockDim.x + threadIdx.x;
  if (i < G3 * GPAD) {
    int r = i >> 7, k = i & (GPAD - 1);
    float v = 0.0f;
    if (k < GIN) v = wih[r * GIN + k] * ihm[r * GIN + k];
    wih_m[i] = v;
  }
  if (i < G3 * Hc) whh_h[i] = (_Float16)(whh[i] * hhm[i]);
}

// ---------------------------------------------------------------------------
// 7. sequential sparse GRU. 2 blocks x 512 thr (16 waves); block = 16 batches
//    = one WMMA M-tile. Per step: 48 N-tiles x 8 unrolled K-WMMAs.
// ---------------------------------------------------------------------------
__global__ void sparse_gru_scan(const float* __restrict__ gi,      // [B,T,768] incl. bih
                                const _Float16* __restrict__ whh,  // [768,256] masked f16
                                const float* __restrict__ bhh,     // [768]
                                float* __restrict__ out) {         // [B,T,256]
  extern __shared__ char smem[];
  float*    h_f = (float*)smem;                       // 16*256 f32
  _Float16* h_h = (_Float16*)(h_f + 16 * Hc);         // 16*256 f16
  float*    gh  = (float*)(h_h + 16 * Hc);            // 16*768 f32

  const int tid  = threadIdx.x;
  const int b0   = blockIdx.x * 16;
  const int lane = tid & 31, wave = tid >> 5;         // 16 waves
  const int l15  = lane & 15, hf = lane >> 4;

  for (int i = tid; i < 16 * Hc; i += blockDim.x) { h_f[i] = 0.0f; h_h[i] = (_Float16)0.0f; }
  __syncthreads();

  const _Float16* hrow = h_h + l15 * Hc;              // this lane's A row

  for (int t = 0; t < Tc; ++t) {
    for (int nt = wave; nt < 48; nt += 16) {          // 3 tiles per wave
      const int n = nt * 16 + l15;
      const _Float16* wr = whh + (size_t)n * Hc;
      v8f c = {};
      #pragma unroll
      for (int k0 = 0; k0 < Hc; k0 += 32) {
        __builtin_prefetch(wr + k0 + 64, 0, 1);       // global_prefetch_b8
        v8h alo = *(const v8h*)(hrow + k0 + (hf << 4));        // ds b128
        v8h ahi = *(const v8h*)(hrow + k0 + 8 + (hf << 4));    // ds b128
        v16h bw = *(const v16h*)(wr + k0 + (hf << 4));         // 2x global b128
        v16h a;
        #pragma unroll
        for (int e = 0; e < 8; ++e) { a[e] = alo[e]; a[e + 8] = ahi[e]; }
        c = __builtin_amdgcn_wmma_f32_16x16x32_f16(false, a, false, bw,
                                                   (short)0, c, false, false);
      }
      #pragma unroll
      for (int v = 0; v < 8; ++v) gh[(v + (hf << 3)) * G3 + n] = c[v];
    }
    __syncthreads();
    for (int i = tid; i < 16 * Hc; i += blockDim.x) {
      int bl = i >> 8;
      int j  = i & 255;
      int b  = b0 + bl;
      const float* gir = gi + ((size_t)b * Tc + t) * G3;
      float ir = gir[j], iz = gir[Hc + j], inn = gir[2 * Hc + j];
      float hr = gh[bl * G3 + j]          + bhh[j];
      float hz = gh[bl * G3 + Hc + j]     + bhh[Hc + j];
      float hn = gh[bl * G3 + 2 * Hc + j] + bhh[2 * Hc + j];
      float r = sigf(ir + hr);
      float z = sigf(iz + hz);
      float n = tanhf(inn + r * hn);
      float hnew = (1.0f - z) * n + z * h_f[i];
      h_f[i] = hnew;
      h_h[i] = (_Float16)hnew;
      out[((size_t)b * Tc + t) * Hc + j] = hnew;
    }
    __syncthreads();
  }
}

// ---------------------------------------------------------------------------
// 8. softmax over 256 classes, one block per (b,t) row
// ---------------------------------------------------------------------------
__global__ void softmax256(const float* __restrict__ logits, float* __restrict__ probs) {
  __shared__ float red[256];
  int j = threadIdx.x;
  size_t row = blockIdx.x;
  float v = logits[row * 256 + j];
  red[j] = v; __syncthreads();
  for (int s = 128; s > 0; s >>= 1) { if (j < s) red[j] = fmaxf(red[j], red[j + s]); __syncthreads(); }
  float mx = red[0]; __syncthreads();
  float e = __expf(v - mx);
  red[j] = e; __syncthreads();
  for (int s = 128; s > 0; s >>= 1) { if (j < s) red[j] += red[j + s]; __syncthreads(); }
  probs[row * 256 + j] = e / red[0];
}

// ---------------------------------------------------------------------------
// host-side orchestration
// ---------------------------------------------------------------------------
static inline int cdiv(long long a, long long b) { return (int)((a + b - 1) / b); }

extern "C" void kernel_launch(void* const* d_in, const int* in_sizes, int n_in,
                              void* d_out, int out_size, void* d_ws, size_t ws_size,
                              hipStream_t stream) {
  (void)in_sizes; (void)n_in; (void)out_size; (void)ws_size;
  const float* mel      = (const float*)d_in[0];
  const float* f0       = (const float*)d_in[1];
  const float* uv       = (const float*)d_in[2];
  const float* audio    = (const float*)d_in[3];
  const float* f0_w     = (const float*)d_in[4];
  const float* f0_b     = (const float*)d_in[5];
  const float* uv_w     = (const float*)d_in[6];
  const float* uv_b     = (const float*)d_in[7];
  const float* conv1_w  = (const float*)d_in[8];
  const float* conv1_b  = (const float*)d_in[9];
  const float* conv2_w  = (const float*)d_in[10];
  const float* conv2_b  = (const float*)d_in[11];
  const float* gruf_wih = (const float*)d_in[12];
  const float* gruf_whh = (const float*)d_in[13];
  const float* gruf_bih = (const float*)d_in[14];
  const float* gruf_bhh = (const float*)d_in[15];
  const float* grub_wih = (const float*)d_in[16];
  const float* grub_whh = (const float*)d_in[17];
  const float* grub_bih = (const float*)d_in[18];
  const float* grub_bhh = (const float*)d_in[19];
  const float* lin1_w   = (const float*)d_in[20];
  const float* lin1_b   = (const float*)d_in[21];
  const float* lin2_w   = (const float*)d_in[22];
  const float* lin2_b   = (const float*)d_in[23];
  const float* cell_wih = (const float*)d_in[24];
  const float* cell_whh = (const float*)d_in[25];
  const float* cell_bih = (const float*)d_in[26];
  const float* cell_bhh = (const float*)d_in[27];
  const float* coarse_w = (const float*)d_in[28];
  const float* coarse_b = (const float*)d_in[29];
  const float* fine_w   = (const float*)d_in[30];
  const float* fine_b   = (const float*)d_in[31];
  const float* ih_mask  = (const float*)d_in[32];
  const float* hh_mask  = (const float*)d_in[33];

  float* out_coarse = (float*)d_out;
  float* out_fine   = out_coarse + (size_t)BT * 256;

  float* ws = (float*)d_ws;
  size_t o = 0;
  float* cin   = ws + o; o += (size_t)BT * 112;
  float* x1    = ws + o; o += (size_t)BT * 64;
  float* x2    = ws + o; o += (size_t)BT * 64;
  float* xcat  = ws + o; o += (size_t)BT * 64;
  float* lin1o = ws + o; o += (size_t)BT * 64;
  float* cond  = ws + o; o += (size_t)BT * 96;
  float* gin   = ws + o; o += (size_t)BT * GPAD;
  float* gib   = ws + o; o += (size_t)BT * G3;
  float* outs  = ws + o; o += (size_t)BT * Hc;
  float* probs = ws + o; o += (size_t)BT * Hc;
  float* wih_m = ws + o; o += (size_t)G3 * GPAD;
  _Float16* whh_h = (_Float16*)(ws + o);             // 768x256 f16

  const int MY = cdiv(BT, 128);                      // 512 macro-rows

  // 1. embeddings + concat
  embed_concat<<<cdiv(BT, 256), 256, 0, stream>>>(mel, f0, uv, f0_w, f0_b, uv_w, uv_b, cin);
  // 2. conditioning convs
  conv1d_lrelu<<<cdiv((long long)BT * 64, 256), 256, 0, stream>>>(cin, conv1_w, conv1_b, x1, 112, 64);
  conv1d_lrelu<<<cdiv((long long)BT * 64, 256), 256, 0, stream>>>(x1, conv2_w, conv2_b, x2, 64, 64);
  // 3. bidirectional tiny GRU -> xcat
  cond_gru_scan<<<Bc, 32, 0, stream>>>(x2, gruf_wih, gruf_whh, gruf_bih, gruf_bhh, xcat, 0, 0);
  cond_gru_scan<<<Bc, 32, 0, stream>>>(x2, grub_wih, grub_whh, grub_bih, grub_bhh, xcat, 1, 32);
  // 4. lin1 (tanh, K=64, N=64) -> lin2 (tanh, K=64, N=96)
  wmma_gemm_fast<4><<<dim3(64 / 64, MY), 256, 0, stream>>>(
      xcat, 64, xcat, 0, lin1_w, lin1_b, lin1o, BT, 64, 1);
  wmma_gemm_fast<2><<<dim3(96 / 32, MY), 256, 0, stream>>>(
      lin1o, 64, lin1o, 0, lin2_w, lin2_b, cond, BT, 96, 1);
  // 5. masked weights + padded gru_in
  mask_weights<<<cdiv(G3 * Hc, 256), 256, 0, stream>>>(cell_wih, ih_mask, cell_whh, hh_mask, wih_m, whh_h);
  build_gruin<<<cdiv(BT, 256), 256, 0, stream>>>(audio, cond, gin);
  // 6. gi = gru_in @ wih_m^T + bih   (K=128 padded, N=768)
  wmma_gemm_fast<4><<<dim3(G3 / 64, MY), 256, 0, stream>>>(
      gin, GPAD, gin, 0, wih_m, cell_bih, gib, BT, G3, 0);
  // 7. sequential sparse GRU
  {
    size_t lds = (size_t)16 * Hc * 4 + (size_t)16 * Hc * 2 + (size_t)16 * G3 * 4; // 72 KB
    sparse_gru_scan<<<2, 512, lds, stream>>>(gib, whh_h, cell_bhh, outs);
  }
  // 8. coarse logits (K=256, N=256)
  wmma_gemm_fast<4><<<dim3(256 / 64, MY), 256, 0, stream>>>(
      outs, 256, outs, 0, coarse_w, coarse_b, out_coarse, BT, 256, 0);
  // 9. coarse softmax
  softmax256<<<BT, 256, 0, stream>>>(out_coarse, probs);
  // 10. fine logits (K=256+256 aligned split, N=256)
  wmma_gemm_fast<4><<<dim3(256 / 64, MY), 256, 0, stream>>>(
      outs, 256, probs, 256, fine_w, fine_b, out_fine, BT, 256, 0);
}
